// TRANSFORMER_CONV_85057532330406
// MI455X (gfx1250) — matrix operand order repaired and checked by reading, not compiled, not run on hardware
//
#include <hip/hip_runtime.h>
#include <cstddef>
#include <cstdint>

#define NN 20000
#define NE 320000
#define HC 256   // H * C
#define CC 64
#define HH 4

typedef __bf16 bf16_t;
typedef __attribute__((ext_vector_type(16))) __bf16 v16bf;
typedef __attribute__((ext_vector_type(2)))  __bf16 bf16x2;
typedef __attribute__((ext_vector_type(8)))  float   v8f;

// ---------- order-preserving float<->uint for atomic max ----------
__device__ __forceinline__ unsigned f2ord(float f) {
  unsigned u = __float_as_uint(f);
  return (u & 0x80000000u) ? ~u : (u | 0x80000000u);
}
__device__ __forceinline__ float ord2f(unsigned u) {
  return (u & 0x80000000u) ? __uint_as_float(u & 0x7fffffffu) : __uint_as_float(~u);
}

// ---------- f32 -> bf16 conversion ----------
__global__ void k_cvt_bf16(const float* __restrict__ in, bf16_t* __restrict__ out, int n) {
  int i = blockIdx.x * blockDim.x + threadIdx.x;
  if (i < n) out[i] = (bf16_t)in[i];
}

// ---------- fragment loaders (ISA 7.12.2 layouts, wave32) ----------
// A 16x32 bf16: half 0 -> K {0..7, 16..23}, half 1 -> K {8..15, 24..31}
__device__ __forceinline__ v16bf load_a_frag(const bf16_t* __restrict__ aRowK, int half) {
  v16bf a;
#pragma unroll
  for (int t = 0; t < 4; ++t) {
    bf16x2 p0 = *(const bf16x2*)(aRowK + half * 8 + 2 * t);
    a[2 * t]     = p0[0];
    a[2 * t + 1] = p0[1];
    bf16x2 p1 = *(const bf16x2*)(aRowK + 16 + half * 8 + 2 * t);
    a[8 + 2 * t]     = p1[0];
    a[8 + 2 * t + 1] = p1[1];
  }
  return a;
}
// B 32x16 bf16: half 0 -> K {0..7, 16..23}, half 1 -> K {8..15, 24..31} (2 per VGPR)
__device__ __forceinline__ v16bf load_b_frag(const bf16_t* __restrict__ wRowK, int half) {
  v16bf b;
#pragma unroll
  for (int j = 0; j < 8; ++j) {
    bf16x2 p = *(const bf16x2*)(wRowK + half * 8 + (j >> 2) * 16 + 2 * (j & 3));
    b[2 * j]     = p[0];
    b[2 * j + 1] = p[1];
  }
  return b;
}

// ---------- WMMA GEMM: out[M,O] = A[M,K] @ W[O,K]^T + bias ----------
// One wave32 computes 16(M) x 64(N). Software-pipelined K-loop: fragment loads
// for step kk+32 are issued before the 4 wmma issues for step kk, so the
// matrix pipe overlaps the 10 outstanding global_load_b128 per step.
__global__ void k_gemm_wmma(const bf16_t* __restrict__ A,    // M x K row-major (bf16)
                            const bf16_t* __restrict__ W,    // O x K row-major (bf16); B[k,n] = W[n*K+k]
                            const float*  __restrict__ bias, // O (nullable)
                            float* __restrict__ out,         // M x O
                            int M, int K, int O)
{
  const int lane  = threadIdx.x & 31;
  const int wave  = threadIdx.x >> 5;
  const int mTile = blockIdx.x * 4 + wave;
  if (mTile * 16 >= M) return;
  const int nBase = blockIdx.y * 64;
  const int half  = lane >> 4;
  const int l16   = lane & 15;
  const int mRow  = mTile * 16 + l16;     // A row held by this lane

  const bf16_t* aRow  = A + (size_t)mRow * K;
  const bf16_t* wRow0 = W + (size_t)(nBase + l16) * K;   // sub-tile n adds 16*K

  v8f acc0, acc1, acc2, acc3;
  {
    float b0 = bias ? bias[nBase + 0 * 16 + l16] : 0.0f;
    float b1 = bias ? bias[nBase + 1 * 16 + l16] : 0.0f;
    float b2 = bias ? bias[nBase + 2 * 16 + l16] : 0.0f;
    float b3 = bias ? bias[nBase + 3 * 16 + l16] : 0.0f;
#pragma unroll
    for (int r = 0; r < 8; ++r) { acc0[r] = b0; acc1[r] = b1; acc2[r] = b2; acc3[r] = b3; }
  }

  // prologue: fragments for kk = 0
  v16bf a  = load_a_frag(aRow, half);
  v16bf b0 = load_b_frag(wRow0 + (size_t)0 * 16 * K, half);
  v16bf b1 = load_b_frag(wRow0 + (size_t)1 * 16 * K, half);
  v16bf b2 = load_b_frag(wRow0 + (size_t)2 * 16 * K, half);
  v16bf b3 = load_b_frag(wRow0 + (size_t)3 * 16 * K, half);

  for (int kk = 32; kk < K; kk += 32) {
    // issue next step's loads first (double buffer)
    v16bf an  = load_a_frag(aRow + kk, half);
    v16bf bn0 = load_b_frag(wRow0 + (size_t)0 * 16 * K + kk, half);
    v16bf bn1 = load_b_frag(wRow0 + (size_t)1 * 16 * K + kk, half);
    v16bf bn2 = load_b_frag(wRow0 + (size_t)2 * 16 * K + kk, half);
    v16bf bn3 = load_b_frag(wRow0 + (size_t)3 * 16 * K + kk, half);
    // compute on resident fragments
    acc0 = __builtin_amdgcn_wmma_f32_16x16x32_bf16(false, a, false, b0, (short)0, acc0, false, false);
    acc1 = __builtin_amdgcn_wmma_f32_16x16x32_bf16(false, a, false, b1, (short)0, acc1, false, false);
    acc2 = __builtin_amdgcn_wmma_f32_16x16x32_bf16(false, a, false, b2, (short)0, acc2, false, false);
    acc3 = __builtin_amdgcn_wmma_f32_16x16x32_bf16(false, a, false, b3, (short)0, acc3, false, false);
    a = an; b0 = bn0; b1 = bn1; b2 = bn2; b3 = bn3;
  }
  // epilogue
  acc0 = __builtin_amdgcn_wmma_f32_16x16x32_bf16(false, a, false, b0, (short)0, acc0, false, false);
  acc1 = __builtin_amdgcn_wmma_f32_16x16x32_bf16(false, a, false, b1, (short)0, acc1, false, false);
  acc2 = __builtin_amdgcn_wmma_f32_16x16x32_bf16(false, a, false, b2, (short)0, acc2, false, false);
  acc3 = __builtin_amdgcn_wmma_f32_16x16x32_bf16(false, a, false, b3, (short)0, acc3, false, false);

  // C/D layout: VGPR r -> row (r + 8*half), column (nBase + sub*16 + l16)
  float* oRow = out + (size_t)(mTile * 16 + half * 8) * O + nBase + l16;
#pragma unroll
  for (int r = 0; r < 8; ++r) {
    oRow[(size_t)r * O + 0]  = acc0[r];
    oRow[(size_t)r * O + 16] = acc1[r];
    oRow[(size_t)r * O + 32] = acc2[r];
    oRow[(size_t)r * O + 48] = acc3[r];
  }
}

// ---------- per-layer init of reduction buffers ----------
__global__ void k_init(unsigned* __restrict__ amax, float* __restrict__ denom,
                       float* __restrict__ segout, int nA, int nS)
{
  int i = blockIdx.x * blockDim.x + threadIdx.x;
  if (i < nA) { amax[i] = 0u; denom[i] = 0.0f; }
  if (i < nS) segout[i] = 0.0f;
}

// ---------- pass A: alpha = <q[dst], k[src]+We@ea>/8 ; atomic segment max ----------
template <bool HAS_E>
__global__ void k_edge_alpha(const float* __restrict__ q, const float* __restrict__ k,
                             const int* __restrict__ src, const int* __restrict__ dst,
                             const float* __restrict__ ea, const float* __restrict__ We,
                             float* __restrict__ alpha, unsigned* __restrict__ amax, int E)
{
  __shared__ float sWe[HAS_E ? HC * 16 : 1];
  if (HAS_E) {
    for (int i = threadIdx.x; i < HC * 16; i += blockDim.x) sWe[i] = We[i];
    __syncthreads();
  }
  int t = blockIdx.x * blockDim.x + threadIdx.x;
  if (t >= E * HH) return;
  int e = t >> 2, h = t & 3;
  int s = src[e], d = dst[e];

  const float4* qp = (const float4*)(q + (size_t)d * HC + h * CC);
  const float4* kp = (const float4*)(k + (size_t)s * HC + h * CC);
  __builtin_prefetch(qp, 0, 3);   // global_prefetch_b8: pull gathered rows toward L0
  __builtin_prefetch(kp, 0, 3);

  float eav[16];
  if (HAS_E) {
#pragma unroll
    for (int j = 0; j < 16; ++j) eav[j] = ea[(size_t)e * 16 + j];
  }

  float acc = 0.0f;
#pragma unroll 4
  for (int c4 = 0; c4 < CC / 4; ++c4) {
    float4 qq = qp[c4];
    float4 kv = kp[c4];
    if (HAS_E) {
      float add[4];
#pragma unroll
      for (int r = 0; r < 4; ++r) {
        const float* wr = &sWe[(h * CC + c4 * 4 + r) * 16];
        float s2 = 0.0f;
#pragma unroll
        for (int j = 0; j < 16; ++j) s2 += eav[j] * wr[j];
        add[r] = s2;
      }
      kv.x += add[0]; kv.y += add[1]; kv.z += add[2]; kv.w += add[3];
    }
    acc += qq.x * kv.x + qq.y * kv.y + qq.z * kv.z + qq.w * kv.w;
  }
  float al = acc * 0.125f;   // 1/sqrt(64)
  alpha[t] = al;
  atomicMax(&amax[(size_t)d * HH + h], f2ord(al));
}

// ---------- pass B: ex = exp(alpha - max); atomic segment sum ----------
__global__ void k_edge_expsum(float* __restrict__ alpha, const int* __restrict__ dst,
                              const unsigned* __restrict__ amax, float* __restrict__ denom, int E)
{
  int t = blockIdx.x * blockDim.x + threadIdx.x;
  if (t >= E * HH) return;
  int e = t >> 2, h = t & 3;
  int d = dst[e];
  float mx = ord2f(amax[(size_t)d * HH + h]);
  float ex = __expf(alpha[t] - mx);
  alpha[t] = ex;
  atomicAdd(&denom[(size_t)d * HH + h], ex);
}

// ---------- pass C: scatter-add (v[src]+We@ea) * alpha/denom into segout ----------
template <bool HAS_E>
__global__ void k_edge_scatter(const float* __restrict__ v, const int* __restrict__ src,
                               const int* __restrict__ dst, const float* __restrict__ ea,
                               const float* __restrict__ We, const float* __restrict__ alpha,
                               const float* __restrict__ denom, float* __restrict__ segout, int E)
{
  __shared__ float sWe[HAS_E ? HC * 16 : 1];
  if (HAS_E) {
    for (int i = threadIdx.x; i < HC * 16; i += blockDim.x) sWe[i] = We[i];
    __syncthreads();
  }
  int t = blockIdx.x * blockDim.x + threadIdx.x;
  if (t >= E * HH) return;
  int e = t >> 2, h = t & 3;
  int s = src[e], d = dst[e];
  float w = alpha[t] / (denom[(size_t)d * HH + h] + 1e-16f);

  const float4* vp = (const float4*)(v + (size_t)s * HC + h * CC);
  float* op = segout + (size_t)d * HC + h * CC;
  __builtin_prefetch(vp, 0, 3);

  float eav[16];
  if (HAS_E) {
#pragma unroll
    for (int j = 0; j < 16; ++j) eav[j] = ea[(size_t)e * 16 + j];
  }

#pragma unroll 4
  for (int c4 = 0; c4 < CC / 4; ++c4) {
    float4 vv = vp[c4];
    if (HAS_E) {
      float add[4];
#pragma unroll
      for (int r = 0; r < 4; ++r) {
        const float* wr = &sWe[(h * CC + c4 * 4 + r) * 16];
        float s2 = 0.0f;
#pragma unroll
        for (int j = 0; j < 16; ++j) s2 += eav[j] * wr[j];
        add[r] = s2;
      }
      vv.x += add[0]; vv.y += add[1]; vv.z += add[2]; vv.w += add[3];
    }
    atomicAdd(&op[c4 * 4 + 0], vv.x * w);
    atomicAdd(&op[c4 * 4 + 1], vv.y * w);
    atomicAdd(&op[c4 * 4 + 2], vv.z * w);
    atomicAdd(&op[c4 * 4 + 3], vv.w * w);
  }
}

// ---------- finalize: beta gate + skip + leaky relu; one wave32 per node ----------
template <bool CONCAT>
__global__ void k_finalize(const float* __restrict__ segout, const float* __restrict__ xr,
                           const float* __restrict__ wbeta, bf16_t* __restrict__ xb_next,
                           float* __restrict__ fout, int N)
{
  const int SO = CONCAT ? HC : CC;
  const int IT = CONCAT ? 8 : 2;   // SO / 32 lanes
  int lane = threadIdx.x & 31;
  int wave = threadIdx.x >> 5;
  int n = blockIdx.x * (blockDim.x >> 5) + wave;
  if (n >= N) return;

  float o[8], xv[8];
  float s = 0.0f;
#pragma unroll
  for (int it = 0; it < IT; ++it) {
    int i = lane + it * 32;
    float ov;
    if (CONCAT) {
      ov = segout[(size_t)n * HC + i];
    } else {
      const float* sp = segout + (size_t)n * HC;
      ov = 0.25f * (sp[i] + sp[CC + i] + sp[2 * CC + i] + sp[3 * CC + i]);
    }
    float xrv = xr[(size_t)n * SO + i];
    o[it] = ov; xv[it] = xrv;
    s += wbeta[i] * ov + wbeta[SO + i] * xrv + wbeta[2 * SO + i] * (ov - xrv);
  }
#pragma unroll
  for (int m = 16; m >= 1; m >>= 1) s += __shfl_xor(s, m, 32);
  float beta = 1.0f / (1.0f + __expf(-s));
#pragma unroll
  for (int it = 0; it < IT; ++it) {
    int i = lane + it * 32;
    float y = beta * xv[it] + (1.0f - beta) * o[it];
    y = (y > 0.0f) ? y : 0.1f * y;                  // leaky relu 0.1
    if (xb_next) xb_next[(size_t)n * SO + i] = (bf16_t)y;
    if (fout)    fout[(size_t)n * SO + i] = y;
  }
}

// ===================== host orchestration =====================
extern "C" void kernel_launch(void* const* d_in, const int* in_sizes, int n_in,
                              void* d_out, int out_size, void* d_ws, size_t ws_size,
                              hipStream_t stream)
{
  const float* x  = (const float*)d_in[0];
  const int*   ei = (const int*)d_in[1];
  const float* ea = (const float*)d_in[2];
  const int* src = ei;
  const int* dst = ei + NE;

  struct LayerP {
    const float *Wq, *bq, *Wk, *bk, *Wv, *bv, *We, *Wskip, *bskip, *Wbeta;
  } L[5];
  int idx = 3;
  for (int l = 0; l < 5; ++l) {
    bool he = (l < 3);
    L[l].Wq    = (const float*)d_in[idx++];
    L[l].bq    = (const float*)d_in[idx++];
    L[l].Wk    = (const float*)d_in[idx++];
    L[l].bk    = (const float*)d_in[idx++];
    L[l].Wv    = (const float*)d_in[idx++];
    L[l].bv    = (const float*)d_in[idx++];
    L[l].We    = he ? (const float*)d_in[idx++] : nullptr;
    L[l].Wskip = (const float*)d_in[idx++];
    L[l].bskip = (const float*)d_in[idx++];
    L[l].Wbeta = (const float*)d_in[idx++];
  }

  // carve workspace (all offsets 256B aligned)
  char* wp = (char*)d_ws;
  auto carve = [&](size_t bytes) -> void* {
    void* p = (void*)wp;
    wp += (bytes + 255) & ~(size_t)255;
    return p;
  };
  bf16_t* xbA  = (bf16_t*)carve((size_t)NN * HC * 2);
  bf16_t* xbB  = (bf16_t*)carve((size_t)NN * HC * 2);
  bf16_t* wqb  = (bf16_t*)carve((size_t)HC * HC * 2);
  bf16_t* wkb  = (bf16_t*)carve((size_t)HC * HC * 2);
  bf16_t* wvb  = (bf16_t*)carve((size_t)HC * HC * 2);
  bf16_t* wsb  = (bf16_t*)carve((size_t)HC * HC * 2);
  float*  q    = (float*)carve((size_t)NN * HC * 4);
  float*  k    = (float*)carve((size_t)NN * HC * 4);
  float*  v    = (float*)carve((size_t)NN * HC * 4);
  float*  xr   = (float*)carve((size_t)NN * HC * 4);
  float*  seg  = (float*)carve((size_t)NN * HC * 4);
  float*  alph = (float*)carve((size_t)NE * HH * 4);
  unsigned* amax = (unsigned*)carve((size_t)NN * HH * 4);
  float*  denom = (float*)carve((size_t)NN * HH * 4);

  auto cvt = [&](const float* in, bf16_t* out, int n) {
    k_cvt_bf16<<<(n + 255) / 256, 256, 0, stream>>>(in, out, n);
  };

  const int mBlocks = (NN / 16 + 3) / 4;   // 1250 M-tiles, 4 waves per block
  const int eThreads = NE * HH;
  const int eBlocks = (eThreads + 255) / 256;

  // initial activation to bf16
  cvt(x, xbA, NN * 64);

  bf16_t* xb = xbA;
  bf16_t* xbn = xbB;
  for (int l = 0; l < 5; ++l) {
    const int K  = (l == 0) ? 64 : HC;
    const int SO = (l == 4) ? CC : HC;

    cvt(L[l].Wq, wqb, HC * K);
    cvt(L[l].Wk, wkb, HC * K);
    cvt(L[l].Wv, wvb, HC * K);
    cvt(L[l].Wskip, wsb, SO * K);

    k_gemm_wmma<<<dim3(mBlocks, HC / 64), 128, 0, stream>>>(xb, wqb, L[l].bq, q, NN, K, HC);
    k_gemm_wmma<<<dim3(mBlocks, HC / 64), 128, 0, stream>>>(xb, wkb, L[l].bk, k, NN, K, HC);
    k_gemm_wmma<<<dim3(mBlocks, HC / 64), 128, 0, stream>>>(xb, wvb, L[l].bv, v, NN, K, HC);
    k_gemm_wmma<<<dim3(mBlocks, SO / 64), 128, 0, stream>>>(xb, wsb, L[l].bskip, xr, NN, K, SO);

    k_init<<<(NN * HC + 255) / 256, 256, 0, stream>>>(amax, denom, seg, NN * HH, NN * HC);

    if (l < 3) {
      k_edge_alpha<true><<<eBlocks, 256, 0, stream>>>(q, k, src, dst, ea, L[l].We,
                                                      alph, amax, NE);
    } else {
      k_edge_alpha<false><<<eBlocks, 256, 0, stream>>>(q, k, src, dst, nullptr, nullptr,
                                                       alph, amax, NE);
    }
    k_edge_expsum<<<eBlocks, 256, 0, stream>>>(alph, dst, amax, denom, NE);
    if (l < 3) {
      k_edge_scatter<true><<<eBlocks, 256, 0, stream>>>(v, src, dst, ea, L[l].We,
                                                        alph, denom, seg, NE);
    } else {
      k_edge_scatter<false><<<eBlocks, 256, 0, stream>>>(v, src, dst, nullptr, nullptr,
                                                         alph, denom, seg, NE);
    }

    if (l < 4) {
      k_finalize<true><<<(NN + 7) / 8, 256, 0, stream>>>(seg, xr, L[l].Wbeta, xbn, nullptr, NN);
      bf16_t* tmp = xb; xb = xbn; xbn = tmp;
    } else {
      k_finalize<false><<<(NN + 7) / 8, 256, 0, stream>>>(seg, xr, L[l].Wbeta, nullptr,
                                                          (float*)d_out, NN);
    }
  }
  (void)in_sizes; (void)n_in; (void)out_size; (void)ws_size;
}
